// LayerAtten_14654428414369
// MI455X (gfx1250) — matrix operand order repaired and checked by reading, not compiled, not run on hardware
//
#include <hip/hip_runtime.h>
#include <stdint.h>

// Problem constants (from reference)
#define B_  16384
#define M_  50
#define D_  128
#define H_  128
#define MP  64            // M padded to 4 tiles of 16
#define KF32ROW 132       // LDS row stride for f32 k tile (128 + 4 pad dwords via TDM pad)

typedef __bf16 bf16_t;
typedef __attribute__((ext_vector_type(16))) __bf16 v16bf;
typedef __attribute__((ext_vector_type(8)))  float  v8f;
typedef unsigned int u32x4 __attribute__((ext_vector_type(4)));
typedef int          i32x8 __attribute__((ext_vector_type(8)));
typedef int          i32x4 __attribute__((ext_vector_type(4)));

union frag_u { uint4 u[2]; v16bf v; };

// B fragment (16-bit B 32x16, ISA layout): 8 bf16 at p and 8 bf16 at p+16
__device__ __forceinline__ v16bf load_fragB(const bf16_t* p) {
    frag_u f;
    f.u[0] = *(const uint4*)p;
    f.u[1] = *(const uint4*)(p + 16);
    return f.v;
}
// Build an A fragment (16-bit A 16x32, ISA layout) from f32 source:
// 8 f32 at p (K = base+lh*8+[0..7]) and 8 f32 at p+16 (K = base+16+lh*8+[0..7])
__device__ __forceinline__ v16bf cvt_fragA(const float* p) {
    union { bf16_t e[16]; v16bf v; } f;
    #pragma unroll
    for (int j = 0; j < 8; ++j) { f.e[j] = (bf16_t)p[j]; f.e[8 + j] = (bf16_t)p[16 + j]; }
    return f.v;
}

// ---- Prologue 1: split W1 into bf16 halves (+ optional f32 transpose of q-half) ----
__global__ __launch_bounds__(256)
void w1_split(const float* __restrict__ W1,
              bf16_t* __restrict__ w1qbf,   // [H][128]  (q-half)
              bf16_t* __restrict__ w1kbf,   // [H][128]  (k-half)
              float*  __restrict__ w1qT) {  // [128][H]  (fallback path only; may be null)
    int i = blockIdx.x * blockDim.x + threadIdx.x;   // 0 .. H*2D-1 (32768)
    int h = i >> 8;
    int c = i & 255;
    float val = W1[i];
    if (c < D_) {
        w1qbf[h * D_ + c] = (bf16_t)val;
        if (w1qT) w1qT[c * H_ + h] = val;
    } else {
        w1kbf[h * D_ + (c - D_)] = (bf16_t)val;
    }
}

// ---- Prologue 2: hq_all[B,H] = q[B,D] @ W1q^T via WMMA ----
__global__ __launch_bounds__(256)
void hq_gemm(const float* __restrict__ q,
             const bf16_t* __restrict__ w1qbf,
             float* __restrict__ hq_all) {
    const int tid  = threadIdx.x;
    const int lane = tid & 31;
    const int wave = tid >> 5;                 // 0..7
    const int ln   = lane & 15;
    const int lh   = lane >> 4;
    const int mbase = blockIdx.x * 128 + wave * 16;
    const float* qrow = q + (size_t)(mbase + ln) * D_;   // this lane's A row

    v8f acc[8] = {};
    #pragma unroll
    for (int kk = 0; kk < 4; ++kk) {
        v16bf a = cvt_fragA(qrow + kk * 32 + lh * 8);
        #pragma unroll
        for (int nt = 0; nt < 8; ++nt) {
            v16bf bfr = load_fragB(w1qbf + (nt * 16 + ln) * D_ + kk * 32 + lh * 8);
            acc[nt] = __builtin_amdgcn_wmma_f32_16x16x32_bf16(
                false, a, false, bfr, (short)0, acc[nt], false, false);
        }
    }
    #pragma unroll
    for (int nt = 0; nt < 8; ++nt) {
        #pragma unroll
        for (int r = 0; r < 8; ++r) {
            int m = mbase + lh * 8 + r;
            hq_all[(size_t)m * H_ + nt * 16 + ln] = acc[nt][r];
        }
    }
}

// ---- Main kernel: one block (4 waves) per batch row ----
__global__ __launch_bounds__(128)
void attn_kernel(const float* __restrict__ k,
                 const float* __restrict__ v,
                 const float* __restrict__ W2,
                 const bf16_t* __restrict__ w1kbf,
                 const float* __restrict__ hq_all,   // preferred path (may be null)
                 const float* __restrict__ q,        // fallback path
                 const float* __restrict__ w1qT,     // fallback path
                 float* __restrict__ out) {
    __shared__ float kf32[MP * KF32ROW];   // TDM destination: k[b] f32, padded rows (33.8 KB)
    __shared__ float qs[D_];
    __shared__ float hq[H_];
    __shared__ float w2s[H_];
    __shared__ float sc[MP];
    __shared__ float attn_s[MP];

    const int b    = blockIdx.x;
    const int tid  = threadIdx.x;        // 0..127
    const int lane = tid & 31;           // wave32
    const int wave = tid >> 5;           // 0..3
    const int ln   = lane & 15;
    const int lh   = lane >> 4;

    const float* kb = k + (size_t)b * M_ * D_;
    const float* vb = v + (size_t)b * M_ * D_;

    // ---- Tensor Data Mover: DMA the 50x128 f32 k-tile into LDS (one instr, TENSORcnt) ----
    if (wave == 0) {
        uint32_t lds_off = (uint32_t)(size_t)(void*)kf32;     // LDS byte offset
        uint64_t ga = (uint64_t)(const void*)kb;              // global tile start
        // D# group0: count=1 | lds_addr | global_addr[56:0] | type=2
        u32x4 g0 = { 1u, lds_off, (uint32_t)ga, (uint32_t)(ga >> 32) | 0x80000000u };
        // D# group1: data_size=2 (4B), pad_enable, pad_interval=6 (128 dw), pad_amount=3 (4 dw),
        //            tensor_dim0=128, tensor_dim1=50, tile_dim0=128, tile_dim1=50, dim0_stride=128
        i32x8 g1 = { (int)0x07920000u,
                     (int)(128u << 16),    // tensor_dim0 low bits [63:48]
                     (int)(50u << 16),     // tensor_dim1 low bits [111:96... 80+]
                     (int)(128u << 16),    // tile_dim0 [127:112]
                     50,                   // tile_dim1 [143:128]
                     128,                  // tensor_dim0_stride [207:160]
                     0, 0 };
        i32x4 gz4 = { 0, 0, 0, 0 };
        i32x8 gz8 = { 0, 0, 0, 0, 0, 0, 0, 0 };
        __builtin_amdgcn_tensor_load_to_lds(g0, g1, gz4, gz4, gz8, 0);
    }

    // Warm the cache path for the v stream we consume at the end.
    __builtin_prefetch(vb + tid * 50, 0, 0);

    // ---- staging (overlapped with TDM) ----
    w2s[tid] = W2[tid];
    if (tid < MP) sc[tid] = 0.0f;
    if (hq_all) hq[tid] = hq_all[(size_t)b * H_ + tid];
    else        qs[tid] = q[(size_t)b * D_ + tid];

    // zero the padded rows 50..63 (contiguous dword span with the row padding)
    {
        uint4 z; z.x = 0u; z.y = 0u; z.z = 0u; z.w = 0u;
        uint4* kz = (uint4*)kf32;
        for (int i = tid; i < (MP - M_) * (KF32ROW / 4); i += 128)
            kz[M_ * (KF32ROW / 4) + i] = z;
    }

    if (wave == 0) __builtin_amdgcn_s_wait_tensorcnt(0);
    __syncthreads();   // publishes TDM-written LDS to all waves

    if (!hq_all) {   // coalesced fallback: hq[h] = q . W1q[h,:] via transposed W1q
        float s = 0.0f;
        #pragma unroll 8
        for (int d = 0; d < D_; ++d) s += qs[d] * w1qT[d * H_ + tid];
        hq[tid] = s;
        __syncthreads();
    }

    // ---- WMMA GEMM: hiddenK[64,128] = bf16(k)[64,128] @ W1k^T[128,128] ----
    // wave owns N-tiles {2*wave, 2*wave+1}; 4 K-steps of 32; 4 M-tiles.
    v8f acc[2][4] = {};
    #pragma unroll
    for (int kk = 0; kk < 4; ++kk) {
        v16bf bfrag[2];
        #pragma unroll
        for (int t = 0; t < 2; ++t) {
            int n = (wave * 2 + t) * 16 + ln;
            bfrag[t] = load_fragB(w1kbf + n * D_ + kk * 32 + lh * 8);
        }
        #pragma unroll
        for (int mt = 0; mt < 4; ++mt) {
            // A fragment straight from the TDM-loaded f32 tile, cvt to bf16 inline
            v16bf a = cvt_fragA(kf32 + (mt * 16 + ln) * KF32ROW + kk * 32 + lh * 8);
            acc[0][mt] = __builtin_amdgcn_wmma_f32_16x16x32_bf16(
                false, a, false, bfrag[0], (short)0, acc[0][mt], false, false);
            acc[1][mt] = __builtin_amdgcn_wmma_f32_16x16x32_bf16(
                false, a, false, bfrag[1], (short)0, acc[1][mt], false, false);
        }
    }

    // ---- fused epilogue: score[m] += sum_n relu(acc + hq[n]) * W2[n] ----
    {
        const int n0 = (wave * 2 + 0) * 16 + ln;
        const int n1 = (wave * 2 + 1) * 16 + ln;
        const float hq0 = hq[n0], w20 = w2s[n0];
        const float hq1 = hq[n1], w21 = w2s[n1];
        #pragma unroll
        for (int mt = 0; mt < 4; ++mt) {
            #pragma unroll
            for (int r = 0; r < 8; ++r) {
                float p = fmaxf(acc[0][mt][r] + hq0, 0.0f) * w20
                        + fmaxf(acc[1][mt][r] + hq1, 0.0f) * w21;
                // 16-lane reduction; half-wave groups hold different m rows
                p += __shfl_xor(p, 1);
                p += __shfl_xor(p, 2);
                p += __shfl_xor(p, 4);
                p += __shfl_xor(p, 8);
                if (ln == 0) {
                    int m = mt * 16 + lh * 8 + r;
                    atomicAdd(&sc[m], p);             // ds_add_f32
                }
            }
        }
    }
    __syncthreads();

    // ---- softmax over M=50 (wave 0) ----
    if (wave == 0) {
        float s0 = sc[lane];
        float s1 = (lane + 32 < M_) ? sc[lane + 32] : -3.0e38f;
        float mx = fmaxf(s0, s1);
        #pragma unroll
        for (int s = 1; s < 32; s <<= 1) mx = fmaxf(mx, __shfl_xor(mx, s));
        float e0 = __expf(s0 - mx);
        float e1 = (lane + 32 < M_) ? __expf(s1 - mx) : 0.0f;
        float sum = e0 + e1;
        #pragma unroll
        for (int s = 1; s < 32; s <<= 1) sum += __shfl_xor(sum, s);
        float inv = 1.0f / sum;
        attn_s[lane] = e0 * inv;
        if (lane + 32 < M_) attn_s[lane + 32] = e1 * inv;
    }
    __syncthreads();

    // ---- out[b,d] = sum_m attn[m] * v[b,m,d]  (coalesced HBM stream) ----
    float o = 0.0f;
    #pragma unroll 10
    for (int m = 0; m < M_; ++m) o += attn_s[m] * vb[m * D_ + tid];
    out[(size_t)b * D_ + tid] = o;
}

extern "C" void kernel_launch(void* const* d_in, const int* in_sizes, int n_in,
                              void* d_out, int out_size, void* d_ws, size_t ws_size,
                              hipStream_t stream) {
    const float* q  = (const float*)d_in[0];
    const float* k  = (const float*)d_in[1];
    const float* v  = (const float*)d_in[2];
    const float* W1 = (const float*)d_in[3];
    const float* W2 = (const float*)d_in[4];
    float* out = (float*)d_out;

    // Workspace layout
    char* base = (char*)d_ws;
    bf16_t* w1qbf = (bf16_t*)base;                    // 32 KB
    bf16_t* w1kbf = (bf16_t*)(base + 32768);          // 32 KB
    float*  w1qT  = (float*)(base + 65536);           // 64 KB (fallback only)
    float*  hqall = (float*)(base + 131072);          // 8 MB
    const size_t need_hq = 131072 + (size_t)B_ * H_ * sizeof(float);
    const bool use_hq = (ws_size >= need_hq);
    const bool have_qt = (ws_size >= 131072);

    w1_split<<<(H_ * 2 * D_) / 256, 256, 0, stream>>>(
        W1, w1qbf, w1kbf, (!use_hq && have_qt) ? w1qT : nullptr);

    if (use_hq) {
        hq_gemm<<<B_ / 128, 256, 0, stream>>>(q, w1qbf, hqall);
        attn_kernel<<<B_, 128, 0, stream>>>(k, v, W2, w1kbf, hqall, nullptr, nullptr, out);
    } else {
        attn_kernel<<<B_, 128, 0, stream>>>(k, v, W2, w1kbf, nullptr, q, w1qT, out);
    }
}
